// TransformerBlock_11175504904438
// MI455X (gfx1250) — compile-verified, matrix operand order checked by reading, not run on hardware
//
#include <hip/hip_runtime.h>
#include <hip/hip_bf16.h>
#include <cstdint>

typedef __bf16 bf16;
typedef __attribute__((ext_vector_type(16))) __bf16 v16bf;
typedef __attribute__((ext_vector_type(8)))  float  v8f;

// ---------------------------------------------------------------------------
// CDNA5 async global->LDS helpers (ASYNCcnt-tracked, VGPR-bypassing copy).
// ---------------------------------------------------------------------------
__device__ __forceinline__ void async_to_lds_b128(unsigned lds_off, const void* gaddr) {
  asm volatile("global_load_async_to_lds_b128 %0, %1, off"
               :: "v"(lds_off), "v"(gaddr) : "memory");
}
__device__ __forceinline__ void wait_asynccnt_4() {
  asm volatile("s_wait_asynccnt 0x4" ::: "memory");
}
__device__ __forceinline__ void wait_asynccnt_0() {
  asm volatile("s_wait_asynccnt 0x0" ::: "memory");
}

// ---------------------------------------------------------------------------
// fp32 -> bf16 conversion (4 elements / thread)
// ---------------------------------------------------------------------------
__global__ __launch_bounds__(256) void cvt_f32_bf16_kernel(const float* __restrict__ in,
                                                           bf16* __restrict__ out,
                                                           size_t n4) {
  size_t i = (size_t)blockIdx.x * blockDim.x + threadIdx.x;
  if (i < n4) {
    float4 f = reinterpret_cast<const float4*>(in)[i];
    size_t o = i * 4;
    out[o + 0] = (bf16)f.x;
    out[o + 1] = (bf16)f.y;
    out[o + 2] = (bf16)f.z;
    out[o + 3] = (bf16)f.w;
  }
}

// ---------------------------------------------------------------------------
// bf16 GEMM: C[M,N] = A[M,K] @ B[K,N] + bias, fp32 accumulation via WMMA.
// Block tile 128x128, 8 waves (2x4), wave tile 64x32 (4x2 of 16x16), BK=32.
// Double-buffered LDS tiles staged with global_load_async_to_lds_b128 so the
// next tile's copy overlaps the current tile's WMMAs.
// EPI: 0 = store f32, 1 = store bf16, 2 = GELU(erf) -> bf16
// ---------------------------------------------------------------------------
enum { EPI_F32 = 0, EPI_BF16 = 1, EPI_GELU = 2 };

template <int EPI>
__global__ __launch_bounds__(256) void gemm_bf16_kernel(
    const bf16* __restrict__ A, const bf16* __restrict__ B,
    const float* __restrict__ bias, float* __restrict__ Cf,
    bf16* __restrict__ Cb, int M, int N, int K) {
  constexpr int BM = 128, BN = 128, BK = 32;
  __shared__ __align__(16) bf16 As[2][BM * BK];
  __shared__ __align__(16) bf16 Bs[2][BK * BN];

  const int tid = threadIdx.x;
  const int wave = tid >> 5, lane = tid & 31;
  const int wm = wave >> 2, wn = wave & 3;          // 2 x 4 wave grid
  const int m0 = blockIdx.y * BM, n0 = blockIdx.x * BN;
  const int lm = lane & 15, kh = lane >> 4;

  v8f acc[4][2] = {};

  const int arow = tid >> 1, ach = (tid & 1) * 16;  // A: 128 rows x 32 halves
  const int brow = tid >> 3, bch = (tid & 7) * 16;  // B: 32 rows x 128 halves

  // Issue the 4 async b128 copies (2 for A, 2 for B) for one K-tile.
  auto stage = [&](int buf, int k0) {
    unsigned lA = (unsigned)(uintptr_t)&As[buf][arow * BK + ach];
    const bf16* gA = A + (size_t)(m0 + arow) * K + k0 + ach;
    async_to_lds_b128(lA, gA);
    async_to_lds_b128(lA + 16, gA + 8);
    unsigned lB = (unsigned)(uintptr_t)&Bs[buf][brow * BN + bch];
    const bf16* gB = B + (size_t)(k0 + brow) * N + n0 + bch;
    async_to_lds_b128(lB, gB);
    async_to_lds_b128(lB + 16, gB + 8);
  };

  stage(0, 0);
  int buf = 0;
  for (int k0 = 0; k0 < K; k0 += BK, buf ^= 1) {
    const bool has_next = (k0 + BK) < K;
    if (has_next) {
      stage(buf ^ 1, k0 + BK);       // overlap next copy with this tile's math
      wait_asynccnt_4();             // async loads retire in order: current tile done
    } else {
      wait_asynccnt_0();
    }
    __syncthreads();

    // --- build fragments per documented CDNA5 bf16 layouts ---
    v16bf afr[4], bfr[2];
#pragma unroll
    for (int i = 0; i < 4; i++) {
      const bf16* ap = &As[buf][(wm * 64 + i * 16 + lm) * BK];
#pragma unroll
      for (int v = 0; v < 8; v++) {
        int kk = (v < 4 ? 2 * v : 16 + 2 * (v - 4)) + 8 * kh;
        afr[i][2 * v]     = ap[kk];
        afr[i][2 * v + 1] = ap[kk + 1];
      }
    }
#pragma unroll
    for (int j = 0; j < 2; j++) {
      int nb = wn * 32 + j * 16 + lm;
#pragma unroll
      for (int v = 0; v < 8; v++) {
        int kk = (v < 4 ? 2 * v : 16 + 2 * (v - 4)) + 8 * kh;
        bfr[j][2 * v]     = Bs[buf][kk * BN + nb];
        bfr[j][2 * v + 1] = Bs[buf][(kk + 1) * BN + nb];
      }
    }
#pragma unroll
    for (int i = 0; i < 4; i++)
#pragma unroll
      for (int j = 0; j < 2; j++)
        acc[i][j] = __builtin_amdgcn_wmma_f32_16x16x32_bf16(
            false, afr[i], false, bfr[j], (short)0, acc[i][j], false, false);
    __syncthreads();
  }

  // --- epilogue ---
  const int r16 = kh * 8;
#pragma unroll
  for (int i = 0; i < 4; i++) {
#pragma unroll
    for (int j = 0; j < 2; j++) {
      int col = n0 + wn * 32 + j * 16 + lm;
      float bv = bias[col];
      int rowb = m0 + wm * 64 + i * 16 + r16;
#pragma unroll
      for (int r = 0; r < 8; r++) {
        float c = acc[i][j][r] + bv;
        size_t idx = (size_t)(rowb + r) * N + col;
        if constexpr (EPI == EPI_F32) {
          Cf[idx] = c;
        } else if constexpr (EPI == EPI_BF16) {
          Cb[idx] = (bf16)c;
        } else {
          float g = 0.5f * c * (1.0f + erff(c * 0.70710678118654752f));
          Cb[idx] = (bf16)g;
        }
      }
    }
  }
}

// ---------------------------------------------------------------------------
// Flash attention (single head, d = D_MODEL = 1024, causal).
// Workgroup = 16 query rows x 1 batch; 8 waves each own 128 output columns
// (full 1024 kept in registers -> scores computed exactly once).
// Key chunk SC=128: waves cooperatively compute 16x128 score tile (WMMA over
// K=1024), online softmax in LDS, P(bf16) in LDS, then P@V via WMMA (K=32).
// ---------------------------------------------------------------------------
__global__ __launch_bounds__(256) void attn_kernel(
    const bf16* __restrict__ Q, const bf16* __restrict__ K,
    const bf16* __restrict__ V, bf16* __restrict__ O, int T, int D) {
  constexpr int SC = 128;
  __shared__ float Sc[16 * SC];
  __shared__ bf16  Pb[16 * SC];
  __shared__ float m_s[16], l_s[16], al_s[16];

  const int b = blockIdx.y;
  const int t0 = blockIdx.x * 16;
  const size_t base = (size_t)b * T * D;
  const int tid = threadIdx.x, wave = tid >> 5, lane = tid & 31;
  const int lm = lane & 15, kh = lane >> 4;
  const float scale = 0.03125f;  // 1/sqrt(1024)

  if (tid < 16) { m_s[tid] = -INFINITY; l_s[tid] = 0.f; }
  v8f acc[8] = {};
  __syncthreads();

  const int s_end = t0 + 16;
  for (int s0 = 0; s0 < s_end; s0 += SC) {
    // --- 1) scores: wave w computes the 16x16 tile at columns s0+16w ---
    {
      v8f sc = {};
      const bf16* qp = Q + base + (size_t)(t0 + lm) * D;
      const bf16* kp = K + base + (size_t)(s0 + wave * 16 + lm) * D;
      for (int kk = 0; kk < D; kk += 32) {
        v16bf af, bfv;
#pragma unroll
        for (int v = 0; v < 8; v++) {
          int k = (v < 4 ? 2 * v : 16 + 2 * (v - 4)) + 8 * kh + kk;
          af[2 * v]      = qp[k];
          af[2 * v + 1]  = qp[k + 1];
          bfv[2 * v]     = kp[k];
          bfv[2 * v + 1] = kp[k + 1];
        }
        sc = __builtin_amdgcn_wmma_f32_16x16x32_bf16(false, af, false, bfv,
                                                     (short)0, sc, false, false);
      }
      int col = wave * 16 + lm;
#pragma unroll
      for (int r = 0; r < 8; r++) Sc[(r + kh * 8) * SC + col] = sc[r];
    }
    __syncthreads();

    // --- 2) online softmax: wave w handles rows 2w, 2w+1 (4 cols / lane) ---
    for (int rr = 0; rr < 2; rr++) {
      int row = wave * 2 + rr;
      int t_abs = t0 + row;
      float mold = m_s[row];
      float vals[4], mx = -INFINITY;
#pragma unroll
      for (int j = 0; j < 4; j++) {
        int c = lane + 32 * j;
        bool valid = (s0 + c) <= t_abs;
        float v = valid ? Sc[row * SC + c] * scale : -INFINITY;
        vals[j] = v;
        mx = fmaxf(mx, v);
      }
      for (int m = 16; m >= 1; m >>= 1) mx = fmaxf(mx, __shfl_xor(mx, m, 32));
      float mnew = fmaxf(mold, mx);
      float sum = 0.f;
#pragma unroll
      for (int j = 0; j < 4; j++) {
        int c = lane + 32 * j;
        bool valid = (s0 + c) <= t_abs;
        float p = valid ? __expf(vals[j] - mnew) : 0.f;
        Pb[row * SC + c] = (bf16)p;
        sum += p;
      }
      for (int m = 16; m >= 1; m >>= 1) sum += __shfl_xor(sum, m, 32);
      if (lane == 0) {
        float alpha = (mold == -INFINITY) ? 0.f : __expf(mold - mnew);
        l_s[row] = l_s[row] * alpha + sum;
        m_s[row] = mnew;
        al_s[row] = alpha;
      }
    }
    __syncthreads();

    // --- 3) rescale running accumulators ---
#pragma unroll
    for (int r = 0; r < 8; r++) {
      float a = al_s[r + kh * 8];
#pragma unroll
      for (int j = 0; j < 8; j++) acc[j][r] *= a;
    }

    // --- 4) O += P @ V : 4 K-steps of 32, 8 column tiles per wave ---
    for (int s2 = 0; s2 < 4; s2++) {
      int kb = s2 * 32;
      v16bf af;
#pragma unroll
      for (int v = 0; v < 8; v++) {
        int k = (v < 4 ? 2 * v : 16 + 2 * (v - 4)) + 8 * kh + kb;
        af[2 * v]     = Pb[lm * SC + k];
        af[2 * v + 1] = Pb[lm * SC + k + 1];
      }
#pragma unroll
      for (int j = 0; j < 8; j++) {
        int col = wave * 128 + j * 16 + lm;
        v16bf bfv;
#pragma unroll
        for (int v = 0; v < 8; v++) {
          int k = (v < 4 ? 2 * v : 16 + 2 * (v - 4)) + 8 * kh;
          bfv[2 * v]     = V[base + (size_t)(s0 + kb + k) * D + col];
          bfv[2 * v + 1] = V[base + (size_t)(s0 + kb + k + 1) * D + col];
        }
        acc[j] = __builtin_amdgcn_wmma_f32_16x16x32_bf16(false, af, false, bfv,
                                                         (short)0, acc[j], false, false);
      }
    }
    __syncthreads();
  }

  // --- epilogue: divide by l, store bf16 ---
#pragma unroll
  for (int r = 0; r < 8; r++) {
    int row = r + kh * 8;
    float inv = 1.0f / l_s[row];
#pragma unroll
    for (int j = 0; j < 8; j++) {
      int col = wave * 128 + j * 16 + lm;
      O[base + (size_t)(t0 + row) * D + col] = (bf16)(acc[j][r] * inv);
    }
  }
}

// ---------------------------------------------------------------------------
// y = LayerNorm(x + res) * g + b ; one block (256 thr) per row of 1024.
// ---------------------------------------------------------------------------
template <bool WRITE_BF16>
__global__ __launch_bounds__(256) void add_ln_kernel(
    const float* __restrict__ x, const float* __restrict__ res,
    const float* __restrict__ g, const float* __restrict__ be,
    float* __restrict__ out_f, bf16* __restrict__ out_b) {
  constexpr int D = 1024;
  __shared__ float red[8];
  const size_t off = (size_t)blockIdx.x * D;
  const int tid = threadIdx.x, wave = tid >> 5, lane = tid & 31;

  float v[4], s = 0.f;
#pragma unroll
  for (int j = 0; j < 4; j++) {
    int c = tid + 256 * j;
    v[j] = x[off + c] + res[off + c];
    s += v[j];
  }
  for (int m = 16; m >= 1; m >>= 1) s += __shfl_xor(s, m, 32);
  if (lane == 0) red[wave] = s;
  __syncthreads();
  float tot = 0.f;
#pragma unroll
  for (int w = 0; w < 8; w++) tot += red[w];
  float mu = tot * (1.0f / D);

  float d2 = 0.f;
#pragma unroll
  for (int j = 0; j < 4; j++) { float d = v[j] - mu; d2 += d * d; }
  for (int m = 16; m >= 1; m >>= 1) d2 += __shfl_xor(d2, m, 32);
  __syncthreads();
  if (lane == 0) red[wave] = d2;
  __syncthreads();
  float tv = 0.f;
#pragma unroll
  for (int w = 0; w < 8; w++) tv += red[w];
  float rstd = rsqrtf(tv * (1.0f / D) + 1e-5f);

#pragma unroll
  for (int j = 0; j < 4; j++) {
    int c = tid + 256 * j;
    float y = (v[j] - mu) * rstd * g[c] + be[c];
    out_f[off + c] = y;
    if constexpr (WRITE_BF16) out_b[off + c] = (bf16)y;
  }
}

// ---------------------------------------------------------------------------
// Host orchestration
// ---------------------------------------------------------------------------
extern "C" void kernel_launch(void* const* d_in, const int* in_sizes, int n_in,
                              void* d_out, int out_size, void* d_ws, size_t ws_size,
                              hipStream_t stream) {
  (void)in_sizes; (void)n_in; (void)out_size; (void)ws_size;
  const float* X  = (const float*)d_in[0];
  const float* WQ = (const float*)d_in[1];
  const float* bQ = (const float*)d_in[2];
  const float* WK = (const float*)d_in[3];
  const float* bK = (const float*)d_in[4];
  const float* WV = (const float*)d_in[5];
  const float* bV = (const float*)d_in[6];
  const float* WO = (const float*)d_in[7];
  const float* bO = (const float*)d_in[8];
  const float* W1 = (const float*)d_in[9];
  const float* b1 = (const float*)d_in[10];
  const float* W2 = (const float*)d_in[11];
  const float* b2 = (const float*)d_in[12];
  const float* g1 = (const float*)d_in[13];
  const float* be1 = (const float*)d_in[14];
  const float* g2 = (const float*)d_in[15];
  const float* be2 = (const float*)d_in[16];
  float* out = (float*)d_out;

  const int B = 4, T = 2048, D = 1024, DFF = 4096;
  const size_t M = (size_t)B * T;  // 8192 rows

  char* ws = (char*)d_ws;
  size_t o = 0;
  auto alloc = [&](size_t bytes) { size_t r = o; o += (bytes + 255) & ~(size_t)255; return r; };

  bf16* Xb  = (bf16*)(ws + alloc(M * D * 2));
  bf16* WQb = (bf16*)(ws + alloc((size_t)D * D * 2));
  bf16* WKb = (bf16*)(ws + alloc((size_t)D * D * 2));
  bf16* WVb = (bf16*)(ws + alloc((size_t)D * D * 2));
  bf16* WOb = (bf16*)(ws + alloc((size_t)D * D * 2));
  bf16* W1b = (bf16*)(ws + alloc((size_t)D * DFF * 2));
  bf16* W2b = (bf16*)(ws + alloc((size_t)DFF * D * 2));
  bf16* big = (bf16*)(ws + alloc(M * DFF * 2));   // 64MB: Q|K|V|attnO, later FFN1 act
  bf16* Qb  = big;
  bf16* Kb  = big + M * D;
  bf16* Vb  = big + 2 * M * D;
  bf16* AOb = big + 3 * M * D;
  bf16* Gb  = big;                                // reuse after attention+O-proj
  float* Of  = (float*)(ws + alloc(M * D * 4));   // O-proj f32, later H2 f32
  float* H1f = (float*)(ws + alloc(M * D * 4));
  bf16*  H1b = (bf16*)(ws + alloc(M * D * 2));

  auto cvt = [&](const float* src, bf16* dst, size_t n) {
    size_t n4 = n / 4;
    cvt_f32_bf16_kernel<<<dim3((unsigned)((n4 + 255) / 256)), dim3(256), 0, stream>>>(src, dst, n4);
  };
  cvt(X,  Xb,  M * D);
  cvt(WQ, WQb, (size_t)D * D);
  cvt(WK, WKb, (size_t)D * D);
  cvt(WV, WVb, (size_t)D * D);
  cvt(WO, WOb, (size_t)D * D);
  cvt(W1, W1b, (size_t)D * DFF);
  cvt(W2, W2b, (size_t)DFF * D);

  dim3 blk(256);
  dim3 gNN(D / 128, (unsigned)(M / 128));    // N=1024
  dim3 gNF(DFF / 128, (unsigned)(M / 128));  // N=4096

  // Q, K, V projections -> bf16
  gemm_bf16_kernel<EPI_BF16><<<gNN, blk, 0, stream>>>(Xb, WQb, bQ, nullptr, Qb, (int)M, D, D);
  gemm_bf16_kernel<EPI_BF16><<<gNN, blk, 0, stream>>>(Xb, WKb, bK, nullptr, Kb, (int)M, D, D);
  gemm_bf16_kernel<EPI_BF16><<<gNN, blk, 0, stream>>>(Xb, WVb, bV, nullptr, Vb, (int)M, D, D);

  // causal flash attention
  attn_kernel<<<dim3(T / 16, B), blk, 0, stream>>>(Qb, Kb, Vb, AOb, T, D);

  // O projection (f32)
  gemm_bf16_kernel<EPI_F32><<<gNN, blk, 0, stream>>>(AOb, WOb, bO, Of, nullptr, (int)M, D, D);

  // H1 = LN(O + X) -> f32 + bf16
  add_ln_kernel<true><<<dim3((unsigned)M), blk, 0, stream>>>(Of, X, g1, be1, H1f, H1b);

  // FFN1 with fused exact GELU -> bf16 (reuses Q/K/V/attnO region)
  gemm_bf16_kernel<EPI_GELU><<<gNF, blk, 0, stream>>>(H1b, W1b, b1, nullptr, Gb, (int)M, DFF, D);

  // FFN2 -> f32 (reuses Of buffer)
  gemm_bf16_kernel<EPI_F32><<<gNN, blk, 0, stream>>>(Gb, W2b, b2, Of, nullptr, (int)M, D, DFF);

  // out = LN(H2 + H1)
  add_ln_kernel<false><<<dim3((unsigned)M), blk, 0, stream>>>(Of, H1f, g2, be2, out, nullptr);
}
